// Model_16655883174574
// MI455X (gfx1250) — matrix operand order repaired and checked by reading, not compiled, not run on hardware
//
#include <hip/hip_runtime.h>
#include <hip/hip_bf16.h>
#include <stdint.h>

// ---------------------------------------------------------------------------
// 2-layer LSTM (B=64, T=2048, H=128) for MI455X / gfx1250 (wave32, WMMA).
//
//  K0a: xf[t][b][h] = x[b][t]*ff_w[h]+ff_b[h]            (f16, parallel)
//  K0b: convert the 4 LSTM weight matrices to f16        (parallel)
//  K1 : gi = xf @ W_ih0^T + b0     big WMMA GEMM         (parallel)
//  K2 : layer-0 recurrence (persistent single workgroup, W_hh in VGPRs)
//  K1': gi = h1 @ W_ih1^T + b1     big WMMA GEMM         (parallel)
//  K2': layer-1 recurrence + fused masked mean/max/last pooling
//  K3 : [64,384] x [384,7] final linear
// ---------------------------------------------------------------------------

typedef _Float16 v16h __attribute__((ext_vector_type(16)));
typedef _Float16 v8h  __attribute__((ext_vector_type(8)));
typedef float    v8f  __attribute__((ext_vector_type(8)));
typedef float    v4f  __attribute__((ext_vector_type(4)));

#define T_STEPS 2048
#define BATCH   64
#define HID     128
#define GATES   512        // 4*H
#define NCLS    7
#define GPITCH  68         // f32 column-major gate buffer pitch (68 % 64 banks = 4)
#define HPITCH  136        // f16 h buffer pitch

union FragU { v16h v; v8h h[2]; };
union AccU  { v8f v; v4f q[2]; float f[8]; };

__device__ __forceinline__ v16h ld_frag(const _Float16* p0, const _Float16* p1) {
  FragU u;
  u.h[0] = *(const v8h*)p0;          // global/ds b128
  u.h[1] = *(const v8h*)p1;          // global/ds b128
  return u.v;
}

__device__ __forceinline__ float sigm(float x) { return 1.0f / (1.0f + __expf(-x)); }

// ------------------------- K0a: input projection to f16 --------------------
__global__ void k_prep_xf(const float* __restrict__ x, const float* __restrict__ ffw,
                          const float* __restrict__ ffb, _Float16* __restrict__ xf) {
  const unsigned idx = blockIdx.x * 256u + threadIdx.x;     // grid sized exactly
  const unsigned t = idx >> 13;            // /(64*128)
  const unsigned r = idx & 8191u;
  const unsigned b = r >> 7;
  const unsigned h = r & 127u;
  xf[idx] = (_Float16)(x[b * 2048u + t] * ffw[h] + ffb[h]);
}

// ------------------------- K0b: weights f32 -> f16 --------------------------
__global__ void k_prep_w16(const float* __restrict__ w0, const float* __restrict__ w1,
                           const float* __restrict__ w2, const float* __restrict__ w3,
                           _Float16* __restrict__ dst) {
  const unsigned idx = blockIdx.x * 256u + threadIdx.x;     // 4 * 65536
  const unsigned m = idx >> 16;
  const float* s = (m == 0) ? w0 : (m == 1) ? w1 : (m == 2) ? w2 : w3;
  dst[idx] = (_Float16)s[idx & 65535u];
}

// --------- K1: gi[M,512] = A[M,128] @ W^T + bias  (M = T*B = 131072) --------
// Block = 8 waves; block owns one 16-row M tile; wave w owns n-tiles 4w..4w+3.
// Explicit 2-stage software pipeline: fragments for K-slice kk+1 are loaded
// before the WMMAs of slice kk issue, so the MAC chain runs behind partial
// loadcnt waits instead of a full drain per WMMA.
__global__ __launch_bounds__(256, 2) void k_gemm(
    const _Float16* __restrict__ A, const _Float16* __restrict__ W,
    const float* __restrict__ bias, _Float16* __restrict__ out) {
  const int tid  = threadIdx.x;
  const int lane = tid & 31, wave = tid >> 5;
  const int col  = lane & 15, khalf = lane >> 4;
  const size_t mTile = blockIdx.x;

  int n[4];
  v8f acc[4];
#pragma unroll
  for (int j = 0; j < 4; ++j) {
    n[j] = (wave * 4 + j) * 16 + col;
    const float bv = bias[n[j]];
    v8f a;
#pragma unroll
    for (int i = 0; i < 8; ++i) a[i] = bv;
    acc[j] = a;
  }

  // A fragment: lane<16 holds K {0..15}; lane>=16 holds K {16..31}
  const _Float16* arow = A + (mTile * 16 + (size_t)col) * HID;
  const _Float16* wrow[4];
#pragma unroll
  for (int j = 0; j < 4; ++j) wrow[j] = W + (size_t)n[j] * HID;

  // ---- prologue: load K-slice 0 -------------------------------------------
  const int ao = khalf * 16;       // A sub-chunk offset within a 32-K slice
  const int bo = khalf * 16;       // B sub-chunk offset within a 32-K slice
  v16h Acur = ld_frag(arow + ao, arow + ao + 8);
  v16h Bcur[4];
#pragma unroll
  for (int j = 0; j < 4; ++j) Bcur[j] = ld_frag(wrow[j] + bo, wrow[j] + bo + 8);

  // ---- steady state: prefetch slice kk+1, multiply slice kk ---------------
#pragma unroll
  for (int kk = 0; kk < 4; ++kk) {
    v16h Anext;
    v16h Bnext[4];
    if (kk < 3) {
      const int ko = (kk + 1) * 32 + ao;
      const int kb = (kk + 1) * 32 + bo;
      Anext = ld_frag(arow + ko, arow + ko + 8);
#pragma unroll
      for (int j = 0; j < 4; ++j)
        Bnext[j] = ld_frag(wrow[j] + kb, wrow[j] + kb + 8);
    }
#pragma unroll
    for (int j = 0; j < 4; ++j)
      acc[j] = __builtin_amdgcn_wmma_f32_16x16x32_f16(
          false, Acur, false, Bcur[j], (short)0, acc[j], false, false);
    if (kk < 3) {
      Acur = Anext;
#pragma unroll
      for (int j = 0; j < 4; ++j) Bcur[j] = Bnext[j];
    }
  }

  const size_t mbase = mTile * 16 + khalf * 8;    // C layout: vgpr p -> row mbase+p
#pragma unroll
  for (int j = 0; j < 4; ++j) {
    AccU u; u.v = acc[j];
#pragma unroll
    for (int p = 0; p < 8; ++p)
      out[(mbase + p) * GATES + n[j]] = (_Float16)u.f[p];
  }
}

// --------------------- K2: persistent recurrent kernel ----------------------
// 512 threads = 16 waves. Wave w owns n-tiles {2w, 2w+1}; W_hh B-fragments
// live in VGPRs for the whole 2048-step loop. h lives in LDS (f16, WMMA-A
// layout friendly); raw h@Whh^T gate sums go to a column-major f32 LDS buffer.
__global__ __launch_bounds__(512, 1) void k_recur(
    const _Float16* __restrict__ gi, const _Float16* __restrict__ whh,
    _Float16* __restrict__ hout, float* __restrict__ pooled,
    const int* __restrict__ lengths, int doPool) {
  __shared__ float    gates[GATES * GPITCH];   // [n][m] column-major, 139 KB
  __shared__ _Float16 hbuf[BATCH * HPITCH];    // [b][k] row-major,     17 KB

  const int tid   = threadIdx.x;
  const int lane  = tid & 31;
  const int wave  = tid >> 5;
  const int col   = lane & 15;
  const int khalf = lane >> 4;

  // ---- preload W_hh B-fragments into registers (persist across all steps) --
  v16h Bf0[4], Bf1[4];
#pragma unroll
  for (int kk = 0; kk < 4; ++kk) {
    const int kb = kk * 32 + khalf * 16;
    const _Float16* w0 = whh + (size_t)((wave * 2 + 0) * 16 + col) * HID + kb;
    const _Float16* w1 = whh + (size_t)((wave * 2 + 1) * 16 + col) * HID + kb;
    Bf0[kk] = ld_frag(w0, w0 + 8);
    Bf1[kk] = ld_frag(w1, w1 + 8);
  }

  // ---- h0 = 0 ----
  for (int i = tid; i < BATCH * HPITCH; i += 512) hbuf[i] = (_Float16)0.0f;

  // ---- per-thread element groups: 2 groups of 8 consecutive j in one b ----
  int gb[2], gj0[2], glen[2];
#pragma unroll
  for (int e = 0; e < 2; ++e) {
    const int grp = e * 512 + tid;             // 1024 groups of 8
    gb[e]   = grp >> 4;
    gj0[e]  = (grp & 15) * 8;
    glen[e] = lengths[gb[e]];
  }

  // ---- register prefetch of gi for t=0 (coalesced b128) ----
  v8h giP[2][4];
#pragma unroll
  for (int e = 0; e < 2; ++e) {
    const _Float16* p = gi + (size_t)gb[e] * GATES + gj0[e];
#pragma unroll
    for (int g = 0; g < 4; ++g) giP[e][g] = *(const v8h*)(p + g * HID);
  }

  float cst[16], psum[16], pmax[16], plast[16];
#pragma unroll
  for (int i = 0; i < 16; ++i) {
    cst[i] = 0.f; psum[i] = 0.f; pmax[i] = -__builtin_inff(); plast[i] = 0.f;
  }

  __syncthreads();

  for (int t = 0; t < T_STEPS; ++t) {
    // ----------------- GEMM: gates = h @ W_hh^T (raw, f32) -----------------
#pragma unroll
    for (int m = 0; m < 4; ++m) {
      v8f a0 = {0.f,0.f,0.f,0.f,0.f,0.f,0.f,0.f};
      v8f a1 = {0.f,0.f,0.f,0.f,0.f,0.f,0.f,0.f};
      const _Float16* ar = hbuf + (m * 16 + col) * HPITCH;
#pragma unroll
      for (int kk = 0; kk < 4; ++kk) {
        const int ko = kk * 32 + khalf * 16;
        const v16h Af = ld_frag(ar + ko, ar + ko + 8);
        a0 = __builtin_amdgcn_wmma_f32_16x16x32_f16(false, Af, false, Bf0[kk],
                                                    (short)0, a0, false, false);
        a1 = __builtin_amdgcn_wmma_f32_16x16x32_f16(false, Af, false, Bf1[kk],
                                                    (short)0, a1, false, false);
      }
      const int mrow = m * 16 + khalf * 8;
      const int n0 = (wave * 2 + 0) * 16 + col;
      const int n1 = (wave * 2 + 1) * 16 + col;
      AccU u0; u0.v = a0;
      AccU u1; u1.v = a1;
      *(v4f*)&gates[n0 * GPITCH + mrow]     = u0.q[0];
      *(v4f*)&gates[n0 * GPITCH + mrow + 4] = u0.q[1];
      *(v4f*)&gates[n1 * GPITCH + mrow]     = u1.q[0];
      *(v4f*)&gates[n1 * GPITCH + mrow + 4] = u1.q[1];
    }
    __syncthreads();

    // -------------------------- LSTM cell update ---------------------------
#pragma unroll
    for (int e = 0; e < 2; ++e) {
      const int b = gb[e], j0 = gj0[e], len = glen[e];
      v8h hv8;
#pragma unroll
      for (int q = 0; q < 8; ++q) {
        const int j = j0 + q, idx = e * 8 + q;
        const float xi = (float)giP[e][0][q] + gates[(0 * HID + j) * GPITCH + b];
        const float xf_= (float)giP[e][1][q] + gates[(1 * HID + j) * GPITCH + b];
        const float xg = (float)giP[e][2][q] + gates[(2 * HID + j) * GPITCH + b];
        const float xo = (float)giP[e][3][q] + gates[(3 * HID + j) * GPITCH + b];
        const float cv = sigm(xf_) * cst[idx] + sigm(xi) * tanhf(xg);
        cst[idx] = cv;
        const float hv = sigm(xo) * tanhf(cv);
        hv8[q] = (_Float16)hv;
        if (doPool) {
          if (t < len)      { psum[idx] += hv; pmax[idx] = fmaxf(pmax[idx], hv); }
          if (t == len - 1) { plast[idx] = hv; }
        }
      }
      *(v8h*)&hbuf[b * HPITCH + j0] = hv8;                               // b128
      if (!doPool)
        *(v8h*)(hout + ((size_t)t * BATCH + b) * HID + j0) = hv8;        // b128
    }

    // ---- prefetch gi[t+1] into registers; hint gi[t+8] toward cache -------
    const int tn = (t + 1 < T_STEPS) ? (t + 1) : t;
#pragma unroll
    for (int e = 0; e < 2; ++e) {
      const _Float16* p = gi + ((size_t)tn * BATCH + gb[e]) * GATES + gj0[e];
#pragma unroll
      for (int g = 0; g < 4; ++g) giP[e][g] = *(const v8h*)(p + g * HID);
    }
    const int tp = (t + 8 < T_STEPS) ? (t + 8) : (T_STEPS - 1);
    __builtin_prefetch(gi + (size_t)tp * BATCH * GATES + tid * 64, 0, 1);

    __syncthreads();
  }

  if (doPool) {
#pragma unroll
    for (int e = 0; e < 2; ++e) {
      const int b = gb[e], j0 = gj0[e];
      const float inv = 1.0f / (float)glen[e];
#pragma unroll
      for (int q = 0; q < 8; ++q) {
        const int j = j0 + q, idx = e * 8 + q;
        pooled[b * 384 + j]       = psum[idx] * inv;
        pooled[b * 384 + 128 + j] = pmax[idx];
        pooled[b * 384 + 256 + j] = plast[idx];
      }
    }
  }
}

// --------------------------- K3: final linear -------------------------------
__global__ void k_final(const float* __restrict__ pooled, const float* __restrict__ lw,
                        const float* __restrict__ lb, float* __restrict__ out) {
  const int tid = threadIdx.x;
  if (tid >= BATCH * NCLS) return;
  const int b = tid / NCLS, c = tid % NCLS;
  float s = lb[c];
  for (int k = 0; k < 3 * HID; ++k) s += pooled[b * 384 + k] * lw[c * 384 + k];
  out[b * NCLS + c] = s;
}

// ---------------------------------------------------------------------------
extern "C" void kernel_launch(void* const* d_in, const int* in_sizes, int n_in,
                              void* d_out, int out_size, void* d_ws, size_t ws_size,
                              hipStream_t stream) {
  const float* x    = (const float*)d_in[0];
  const int*   len  = (const int*)d_in[1];
  const float* ffw  = (const float*)d_in[2];
  const float* ffb  = (const float*)d_in[3];
  const float* wih0 = (const float*)d_in[4];
  const float* whh0 = (const float*)d_in[5];
  const float* b0   = (const float*)d_in[6];
  const float* wih1 = (const float*)d_in[7];
  const float* whh1 = (const float*)d_in[8];
  const float* b1   = (const float*)d_in[9];
  const float* lw   = (const float*)d_in[10];
  const float* lb   = (const float*)d_in[11];

  char* ws = (char*)d_ws;
  _Float16* xf     = (_Float16*)(ws + 0);                 //  32 MB  [T][B][H]
  _Float16* gi     = (_Float16*)(ws + 33554432ull);       // 128 MB  [T][B][4H] (reused)
  _Float16* h1     = (_Float16*)(ws + 167772160ull);      //  32 MB  [T][B][H]
  _Float16* w16    = (_Float16*)(ws + 201326592ull);      // 512 KB  4 x [512][128]
  float*    pooled = (float*)  (ws + 201850880ull);       //  96 KB  [64][384]

  // Parallel prep
  k_prep_xf <<<65536, 256, 0, stream>>>(x, ffw, ffb, xf);
  k_prep_w16<<<1024, 256, 0, stream>>>(wih0, whh0, wih1, whh1, w16);

  // Layer 0: parallel input GEMM, then recurrence
  k_gemm <<<8192, 256, 0, stream>>>(xf, w16 + 0 * 65536, b0, gi);
  k_recur<<<1, 512, 0, stream>>>(gi, w16 + 1 * 65536, h1, nullptr, len, 0);

  // Layer 1: parallel input GEMM (gi buffer reused), recurrence + pooling
  k_gemm <<<8192, 256, 0, stream>>>(h1, w16 + 2 * 65536, b1, gi);
  k_recur<<<1, 512, 0, stream>>>(gi, w16 + 3 * 65536, nullptr, pooled, len, 1);

  // Final classifier
  k_final<<<1, 512, 0, stream>>>(pooled, lw, lb, (float*)d_out);
}
